// AttentionBlock_26147760898687
// MI455X (gfx1250) — matrix operand order repaired and checked by reading, not compiled, not run on hardware
//
#include <hip/hip_runtime.h>

// ---------------------------------------------------------------------------
// Types for CDNA5 WMMA (wave32, v_wmma_f32_16x16x32_bf16)
// ---------------------------------------------------------------------------
typedef __attribute__((ext_vector_type(16))) __bf16         v16bf;
typedef __attribute__((ext_vector_type(8)))  float          v8f;
typedef __attribute__((ext_vector_type(8)))  unsigned short ushort8;

union FragU {
    ushort8 h[2];
    v16bf   v;
};

__device__ __forceinline__ unsigned short f32_to_bf16(float f) {
    unsigned int u = __builtin_bit_cast(unsigned int, f);
    u += 0x7FFFu + ((u >> 16) & 1u);          // round-to-nearest-even
    return (unsigned short)(u >> 16);
}

// A fragment: 16x32 bf16, M = lane&15, half = lane>>4.
// VGPR0..3 hold K = half*8 .. half*8+7, VGPR4..7 hold K = 16+half*8 .. +7.
__device__ __forceinline__ v16bf load_frag_a(const unsigned short* __restrict__ base,
                                             int ld, int k0, int lane) {
    int m = lane & 15, half = lane >> 4;
    const unsigned short* p = base + (size_t)m * ld + k0 + half * 8;
    FragU f;
    f.h[0] = *(const ushort8*)(p);
    f.h[1] = *(const ushort8*)(p + 16);
    return f.v;
}

// B fragment: 32x16 bf16, N = lane&15, half = lane>>4, lane holds
// K = half*8 .. half*8+7 and 16+half*8 .. +7 of column N (Bt[n][k] row-major).
__device__ __forceinline__ v16bf load_frag_b(const unsigned short* __restrict__ base,
                                             int ld, int k0, int lane) {
    int n = lane & 15, half = lane >> 4;
    const unsigned short* p = base + (size_t)n * ld + k0 + half * 8;
    FragU f;
    f.h[0] = *(const ushort8*)(p);
    f.h[1] = *(const ushort8*)(p + 16);
    return f.v;
}

#define WMMA_BF16(A, B, C) \
    __builtin_amdgcn_wmma_f32_16x16x32_bf16(false, (A), false, (B), (short)0, (C), false, false)

// Wave tile: 32 (M) x 64 (N) = 2 M-frags x 4 N-frags, double-buffered k-steps.
struct FragSet {
    v16bf a[2];
    v16bf b[4];
};

__device__ __forceinline__ void load_set(FragSet& fs,
                                         const unsigned short* __restrict__ arow0,
                                         const unsigned short* __restrict__ arow1, int lda,
                                         const unsigned short* __restrict__ bbase, int ldb,
                                         int k0, int lane) {
    fs.a[0] = load_frag_a(arow0, lda, k0, lane);
    fs.a[1] = load_frag_a(arow1, lda, k0, lane);
#pragma unroll
    for (int t = 0; t < 4; ++t)
        fs.b[t] = load_frag_b(bbase + (size_t)(16 * t) * ldb, ldb, k0, lane);
}

__device__ __forceinline__ void mma_set(v8f acc[2][4], const FragSet& fs) {
#pragma unroll
    for (int mi = 0; mi < 2; ++mi)
#pragma unroll
        for (int t = 0; t < 4; ++t)
            acc[mi][t] = WMMA_BF16(fs.a[mi], fs.b[t], acc[mi][t]);
}

// ---------------------------------------------------------------------------
// 1) f32 -> bf16 weight conversion
// ---------------------------------------------------------------------------
__global__ __launch_bounds__(256) void cvt_bf16_kernel(const float* __restrict__ src,
                                                       unsigned short* __restrict__ dst, int n) {
    int i = blockIdx.x * 256 + threadIdx.x;
    if (i < n) dst[i] = f32_to_bf16(src[i]);
}

// ---------------------------------------------------------------------------
// 2) GroupNorm statistics: one block per (b, g), 16*1024 = 16384 contiguous f32
// ---------------------------------------------------------------------------
__global__ __launch_bounds__(256) void gn_stats_kernel(const float* __restrict__ x,
                                                       float* __restrict__ mean,
                                                       float* __restrict__ rstd) {
    int bg = blockIdx.x;                       // b*32 + g
    const float* base = x + (size_t)bg * 16384;
    int t = threadIdx.x;
    float s = 0.f, sq = 0.f;
    for (int i = t; i < 16384; i += 256) {
        float v = base[i];
        s += v; sq += v * v;
    }
    __shared__ float rs[256], rq[256];
    rs[t] = s; rq[t] = sq;
    __syncthreads();
    for (int k = 128; k > 0; k >>= 1) {
        if (t < k) { rs[t] += rs[t + k]; rq[t] += rq[t + k]; }
        __syncthreads();
    }
    if (t == 0) {
        float mu  = rs[0] * (1.0f / 16384.0f);
        float var = rq[0] * (1.0f / 16384.0f) - mu * mu;
        mean[bg] = mu;
        rstd[bg] = rsqrtf(var + 1e-5f);
    }
}

// ---------------------------------------------------------------------------
// 3) Normalize + transpose to bf16 [token, channel] (token = b*1024 + hw)
// ---------------------------------------------------------------------------
__global__ __launch_bounds__(256) void gn_apply_kernel(const float* __restrict__ x,
                                                       const float* __restrict__ gamma,
                                                       const float* __restrict__ beta,
                                                       const float* __restrict__ mean,
                                                       const float* __restrict__ rstd,
                                                       unsigned short* __restrict__ hn) {
    int token = blockIdx.x;
    int b = token >> 10, hw = token & 1023;
    for (int c = threadIdx.x; c < 512; c += 256) {
        float v  = x[((size_t)(b * 512 + c)) * 1024 + hw];
        int   bg = b * 32 + (c >> 4);
        float nv = (v - mean[bg]) * rstd[bg] * gamma[c] + beta[c];
        hn[(size_t)token * 512 + c] = f32_to_bf16(nv);
    }
}

// ---------------------------------------------------------------------------
// 4) QKV GEMM: [16384,512] x [512,1536] (+bias). q pre-scaled by 1/sqrt(C),
//    v written transposed vT[b][c][hw] for the P*V GEMM.
//    Block tile 64x256 (8 waves, each 32x64), double-buffered K loop.
// ---------------------------------------------------------------------------
__global__ __launch_bounds__(256) void gemm_qkv_kernel(const unsigned short* __restrict__ hn,
                                                       const unsigned short* __restrict__ w,
                                                       const float* __restrict__ bias,
                                                       unsigned short* __restrict__ qb,
                                                       unsigned short* __restrict__ kb,
                                                       unsigned short* __restrict__ vT) {
    int lane = threadIdx.x & 31, wave = threadIdx.x >> 5;
    int mbase = blockIdx.x * 64 + (wave >> 2) * 32;
    int nbase = blockIdx.y * 256 + (wave & 3) * 64;
    const unsigned short* arow0 = hn + (size_t)mbase * 512;
    const unsigned short* arow1 = arow0 + (size_t)16 * 512;
    const unsigned short* bbase = w + (size_t)nbase * 512;
    v8f acc[2][4] = {};
    FragSet cur, nxt;
    load_set(cur, arow0, arow1, 512, bbase, 512, 0, lane);
    for (int k0 = 32; k0 < 512; k0 += 32) {
        load_set(nxt, arow0, arow1, 512, bbase, 512, k0, lane);
        __builtin_prefetch(arow0 + (lane & 15) * 512 + k0 + 32, 0, 1);
        mma_set(acc, cur);
        cur = nxt;
    }
    mma_set(acc, cur);

    int n = lane & 15, half = lane >> 4;
    const float scale = 0.044194173824159216f; // 1/sqrt(512)
#pragma unroll
    for (int mi = 0; mi < 2; ++mi) {
#pragma unroll
        for (int t = 0; t < 4; ++t) {
            int   o  = nbase + 16 * t + n;
            float bv = bias[o];
#pragma unroll
            for (int r = 0; r < 8; ++r) {
                int   token = mbase + mi * 16 + half * 8 + r;
                float val   = acc[mi][t][r] + bv;
                if (o < 512) {
                    qb[(size_t)token * 512 + o] = f32_to_bf16(val * scale);
                } else if (o < 1024) {
                    kb[(size_t)token * 512 + (o - 512)] = f32_to_bf16(val);
                } else {
                    int b = token >> 10, hw = token & 1023;
                    vT[((size_t)(b * 512 + (o - 1024))) * 1024 + hw] = f32_to_bf16(val);
                }
            }
        }
    }
}

// ---------------------------------------------------------------------------
// 5) Scores GEMM per batch: S[b][i][j] = q[b][i][:] . k[b][j][:]  (f32 out)
// ---------------------------------------------------------------------------
__global__ __launch_bounds__(256) void gemm_scores_kernel(const unsigned short* __restrict__ qb,
                                                          const unsigned short* __restrict__ kb,
                                                          float* __restrict__ S) {
    int lane = threadIdx.x & 31, wave = threadIdx.x >> 5;
    int mbase = blockIdx.x * 64 + (wave >> 2) * 32;    // global token (query)
    int nbase = blockIdx.y * 256 + (wave & 3) * 64;    // key index j
    int batch = mbase >> 10;
    const unsigned short* arow0 = qb + (size_t)mbase * 512;
    const unsigned short* arow1 = arow0 + (size_t)16 * 512;
    const unsigned short* bbase = kb + ((size_t)batch * 1024 + nbase) * 512;
    v8f acc[2][4] = {};
    FragSet cur, nxt;
    load_set(cur, arow0, arow1, 512, bbase, 512, 0, lane);
    for (int k0 = 32; k0 < 512; k0 += 32) {
        load_set(nxt, arow0, arow1, 512, bbase, 512, k0, lane);
        mma_set(acc, cur);
        cur = nxt;
    }
    mma_set(acc, cur);

    int n = lane & 15, half = lane >> 4;
#pragma unroll
    for (int mi = 0; mi < 2; ++mi) {
#pragma unroll
        for (int t = 0; t < 4; ++t) {
            int j = nbase + 16 * t + n;
#pragma unroll
            for (int r = 0; r < 8; ++r) {
                int token = mbase + mi * 16 + half * 8 + r; // = batch*1024 + i
                S[(size_t)token * 1024 + j] = acc[mi][t][r];
            }
        }
    }
}

// ---------------------------------------------------------------------------
// 6) Row softmax, in-place f32 -> bf16 (row pitch stays 4096 B = 2048 ushorts)
// ---------------------------------------------------------------------------
__global__ __launch_bounds__(256) void softmax_kernel(float* __restrict__ S) {
    int    row = blockIdx.x;
    float* rp  = S + (size_t)row * 1024;
    int    t   = threadIdx.x;
    float4 v   = ((const float4*)rp)[t];
    __shared__ float red[256];
    float m = fmaxf(fmaxf(v.x, v.y), fmaxf(v.z, v.w));
    red[t] = m;
    __syncthreads();
    for (int s = 128; s > 0; s >>= 1) {
        if (t < s) red[t] = fmaxf(red[t], red[t + s]);
        __syncthreads();
    }
    float rowmax = red[0];
    __syncthreads();
    float e0 = __expf(v.x - rowmax), e1 = __expf(v.y - rowmax);
    float e2 = __expf(v.z - rowmax), e3 = __expf(v.w - rowmax);
    red[t] = e0 + e1 + e2 + e3;
    __syncthreads();
    for (int s = 128; s > 0; s >>= 1) {
        if (t < s) red[t] += red[t + s];
        __syncthreads();
    }
    float inv = 1.0f / red[0];
    unsigned short* pp = (unsigned short*)rp;         // all f32 reads already done
    pp[4 * t + 0] = f32_to_bf16(e0 * inv);
    pp[4 * t + 1] = f32_to_bf16(e1 * inv);
    pp[4 * t + 2] = f32_to_bf16(e2 * inv);
    pp[4 * t + 3] = f32_to_bf16(e3 * inv);
}

// ---------------------------------------------------------------------------
// 7) P*V GEMM: O[b][i][c] = sum_j P[b][i][j] * vT[b][c][j]   (K = 1024)
// ---------------------------------------------------------------------------
__global__ __launch_bounds__(256) void gemm_attnv_kernel(const unsigned short* __restrict__ P,
                                                         const unsigned short* __restrict__ vT,
                                                         unsigned short* __restrict__ attn_out) {
    int lane = threadIdx.x & 31, wave = threadIdx.x >> 5;
    int mbase = blockIdx.x * 64 + (wave >> 2) * 32;    // global token
    int nbase = blockIdx.y * 256 + (wave & 3) * 64;    // channel c
    int batch = mbase >> 10;
    const unsigned short* arow0 = P + (size_t)mbase * 2048;          // pitch 2048 ushorts
    const unsigned short* arow1 = arow0 + (size_t)16 * 2048;
    const unsigned short* bbase = vT + ((size_t)batch * 512 + nbase) * 1024;
    v8f acc[2][4] = {};
    FragSet cur, nxt;
    load_set(cur, arow0, arow1, 2048, bbase, 1024, 0, lane);
    for (int k0 = 32; k0 < 1024; k0 += 32) {
        load_set(nxt, arow0, arow1, 2048, bbase, 1024, k0, lane);
        mma_set(acc, cur);
        cur = nxt;
    }
    mma_set(acc, cur);

    int n = lane & 15, half = lane >> 4;
#pragma unroll
    for (int mi = 0; mi < 2; ++mi) {
#pragma unroll
        for (int t = 0; t < 4; ++t) {
            int c = nbase + 16 * t + n;
#pragma unroll
            for (int r = 0; r < 8; ++r) {
                int token = mbase + mi * 16 + half * 8 + r;
                attn_out[(size_t)token * 512 + c] = f32_to_bf16(acc[mi][t][r]);
            }
        }
    }
}

// ---------------------------------------------------------------------------
// 8) Projection GEMM + bias + residual -> f32 output [B, C, H, W]
// ---------------------------------------------------------------------------
__global__ __launch_bounds__(256) void gemm_proj_kernel(const unsigned short* __restrict__ ao,
                                                        const unsigned short* __restrict__ w,
                                                        const float* __restrict__ bias,
                                                        const float* __restrict__ x,
                                                        float* __restrict__ out) {
    int lane = threadIdx.x & 31, wave = threadIdx.x >> 5;
    int mbase = blockIdx.x * 64 + (wave >> 2) * 32;
    int nbase = blockIdx.y * 256 + (wave & 3) * 64;
    const unsigned short* arow0 = ao + (size_t)mbase * 512;
    const unsigned short* arow1 = arow0 + (size_t)16 * 512;
    const unsigned short* bbase = w + (size_t)nbase * 512;
    v8f acc[2][4] = {};
    FragSet cur, nxt;
    load_set(cur, arow0, arow1, 512, bbase, 512, 0, lane);
    for (int k0 = 32; k0 < 512; k0 += 32) {
        load_set(nxt, arow0, arow1, 512, bbase, 512, k0, lane);
        mma_set(acc, cur);
        cur = nxt;
    }
    mma_set(acc, cur);

    int n = lane & 15, half = lane >> 4;
#pragma unroll
    for (int mi = 0; mi < 2; ++mi) {
#pragma unroll
        for (int t = 0; t < 4; ++t) {
            int   o  = nbase + 16 * t + n;
            float bv = bias[o];
#pragma unroll
            for (int r = 0; r < 8; ++r) {
                int    token = mbase + mi * 16 + half * 8 + r;
                int    b = token >> 10, hw = token & 1023;
                size_t idx = ((size_t)(b * 512 + o)) * 1024 + hw;
                out[idx] = x[idx] + acc[mi][t][r] + bv;
            }
        }
    }
}

// ---------------------------------------------------------------------------
// Launcher. Workspace layout (bytes):
//   [0)          qkv_w bf16   1536*512*2 = 1572864
//   [1572864)    proj_w bf16   512*512*2 =  524288
//   [2097152)    mean          512*4
//   [2099200)    rstd          512*4
//   [2101248)    hn bf16     16384*512*2 = 16777216
//   [18878464)   q  bf16                  16777216
//   [35655680)   k  bf16                  16777216
//   [52432896)   vT bf16                  16777216
//   [69210112)   S  f32   16*1024*1024*4 = 67108864  (softmax rewrites bf16 P in-place)
//   [136318976)  attn_out bf16            16777216   -> total ~153 MB
// ---------------------------------------------------------------------------
extern "C" void kernel_launch(void* const* d_in, const int* in_sizes, int n_in,
                              void* d_out, int out_size, void* d_ws, size_t ws_size,
                              hipStream_t stream) {
    const float* x      = (const float*)d_in[0];
    const float* gamma  = (const float*)d_in[1];
    const float* beta   = (const float*)d_in[2];
    const float* qkv_w  = (const float*)d_in[3];
    const float* qkv_b  = (const float*)d_in[4];
    const float* proj_w = (const float*)d_in[5];
    const float* proj_b = (const float*)d_in[6];
    float*       out    = (float*)d_out;

    char* ws = (char*)d_ws;
    unsigned short* qkvw_bf  = (unsigned short*)(ws + 0);
    unsigned short* projw_bf = (unsigned short*)(ws + 1572864);
    float*          meanb    = (float*)(ws + 2097152);
    float*          rstdb    = (float*)(ws + 2099200);
    unsigned short* hn       = (unsigned short*)(ws + 2101248);
    unsigned short* qb       = (unsigned short*)(ws + 18878464);
    unsigned short* kb       = (unsigned short*)(ws + 35655680);
    unsigned short* vT       = (unsigned short*)(ws + 52432896);
    float*          S        = (float*)(ws + 69210112);
    unsigned short* attn_out = (unsigned short*)(ws + 136318976);

    cvt_bf16_kernel<<<3072, 256, 0, stream>>>(qkv_w, qkvw_bf, 1536 * 512);
    cvt_bf16_kernel<<<1024, 256, 0, stream>>>(proj_w, projw_bf, 512 * 512);
    gn_stats_kernel<<<512, 256, 0, stream>>>(x, meanb, rstdb);
    gn_apply_kernel<<<16384, 256, 0, stream>>>(x, gamma, beta, meanb, rstdb, hn);
    gemm_qkv_kernel<<<dim3(256, 6), 256, 0, stream>>>(hn, qkvw_bf, qkv_b, qb, kb, vT);
    gemm_scores_kernel<<<dim3(256, 4), 256, 0, stream>>>(qb, kb, S);
    softmax_kernel<<<16384, 256, 0, stream>>>(S);
    gemm_attnv_kernel<<<dim3(256, 2), 256, 0, stream>>>((const unsigned short*)S, vT, attn_out);
    gemm_proj_kernel<<<dim3(256, 2), 256, 0, stream>>>(attn_out, projw_bf, proj_b, x, out);
}
